// FGGA_84447646974563
// MI455X (gfx1250) — compile-verified, matrix operand order checked by reading, not run on hardware
//
#include <hip/hip_runtime.h>
#include <math.h>

typedef __attribute__((ext_vector_type(16))) _Float16 v16h;
typedef __attribute__((ext_vector_type(16))) __bf16   v16bf;
typedef __attribute__((ext_vector_type(8)))  float    v8f;

namespace {

constexpr int kB     = 8;
constexpr int kCin   = 128;
constexpr int kCd    = 64;     // DOWN_DIM
constexpr int kH     = 128;
constexpr int kW     = 128;
constexpr int kN     = kH * kW;  // 16384
constexpr int kHeads = 8;
constexpr int kHd    = 8;
constexpr float kBnEps = 1e-5f;
constexpr int kLdsPitch = 129;                       // +1 pad: conflict-free column reads
constexpr int kLdsBytes = 2 * kH * kLdsPitch * (int)sizeof(float);  // 132 KB < 320 KB/WGP

// ---- ISA 7.12.2: 16-bit A-matrix 16x32 lane/VGPR layout ------------------
// half index i (0..15), lane (0..31) -> K index.
// Note k-runs are contiguous: i=0..7 -> k=off+0..7, i=8..15 -> k=16+off+0..7
// (off = lane>=16 ? 8 : 0), so kStride==1 fragments vectorize to b128 loads.
__device__ __forceinline__ int fragK(int i, int lane) {
  int v = i >> 1;
  return ((v & 4) << 2) | ((lane >> 4) << 3) | ((v & 3) << 1) | (i & 1);
}

// Split-bf16 fragment (hi + lo covers ~16 mantissa bits)
struct Frag { v16bf hi, lo; };

__device__ __forceinline__ Frag make_frag(const float* __restrict__ p,
                                          int laneStride, int kStride, int lane) {
  Frag f;
  int l = lane & 15;
  #pragma unroll
  for (int i = 0; i < 16; ++i) {
    int k = fragK(i, lane);
    float v = p[l * laneStride + k * kStride];
    __bf16 h = (__bf16)v;
    f.hi[i] = h;
    f.lo[i] = (__bf16)(v - (float)h);
  }
  return f;
}

__device__ __forceinline__ Frag fneg(const Frag& a) {
  Frag r;
  #pragma unroll
  for (int i = 0; i < 16; ++i) {
    r.hi[i] = (__bf16)(-(float)a.hi[i]);
    r.lo[i] = (__bf16)(-(float)a.lo[i]);
  }
  return r;
}

// bf16x3 split product: acc += A*B with ~f32 accuracy (3 WMMAs)
__device__ __forceinline__ v8f mma3(const Frag& a, const Frag& b, v8f c) {
  c = __builtin_amdgcn_wmma_f32_16x16x32_bf16(false, a.hi, false, b.hi, (short)0, c, false, false);
  c = __builtin_amdgcn_wmma_f32_16x16x32_bf16(false, a.hi, false, b.lo, (short)0, c, false, false);
  c = __builtin_amdgcn_wmma_f32_16x16x32_bf16(false, a.lo, false, b.hi, (short)0, c, false, false);
  return c;
}

__device__ __forceinline__ float gelu_exact(float x) {
  return 0.5f * x * (1.0f + erff(x * 0.70710678118654752f));
}

// ---- DFT twiddle matrix: W[j][k] = exp(-2*pi*i*j*k/128) (symmetric) ------
__global__ void init_dft(float* __restrict__ Wr, float* __restrict__ Wi) {
  int idx = blockIdx.x * 256 + threadIdx.x;           // 16384 entries
  int j = idx >> 7, k = idx & 127;
  int t = (j * k) & 127;                              // exact period reduction
  float ang = -6.283185307179586f * (float)t / 128.0f;
  Wr[idx] = cosf(ang);
  Wi[idx] = sinf(ang);
}

// ---- Repack conv weights [co][ci][tap] -> [tap][co][ci] for contiguous A -
__global__ void repack_w(const float* __restrict__ wgt, float* __restrict__ wp) {
  int idx = blockIdx.x * 256 + threadIdx.x;           // 9*64*128 = 73728
  if (idx >= 9 * kCd * kCin) return;
  int ci  = idx & 127;
  int co  = (idx >> 7) & 63;
  int tap = idx >> 13;
  wp[((size_t)tap * kCd + co) * kCin + ci] = wgt[((size_t)co * kCin + ci) * 9 + tap];
}

// ---- Kernel 1: dilated 3x3 conv (implicit GEMM, f16 WMMA) + BN + GELU ----
// grid = B*H*(W/16), block = 128 (4 waves, each owns one 16-channel co tile)
__global__ void conv_bn_gelu(const float* __restrict__ x, const float* __restrict__ wp,
                             const float* __restrict__ g, const float* __restrict__ be,
                             const float* __restrict__ mu, const float* __restrict__ va,
                             float* __restrict__ feat) {
  int blk  = blockIdx.x;
  int wt   = blk & 7;
  int h    = (blk >> 3) & 127;
  int b    = blk >> 10;
  int wave = threadIdx.x >> 5;
  int lane = threadIdx.x & 31;
  int co0  = wave * 16;
  int w0   = wt * 16;

  v8f acc = {};
  #pragma unroll
  for (int kh = 0; kh < 3; ++kh) {
    int hh = h + 3 * (kh - 1);
    if (hh < 0 || hh >= kH) continue;               // block-uniform branch
    #pragma unroll
    for (int kw = 0; kw < 3; ++kw) {
      int dw  = 3 * (kw - 1);
      int tap = kh * 3 + kw;
      int n   = lane & 15;
      int wc  = w0 + n + dw;
      int wcc = wc < 0 ? 0 : (wc > kW - 1 ? kW - 1 : wc);   // clamp address
      float msk = (wc == wcc) ? 1.0f : 0.0f;                // zero-pad via select
      for (int kc = 0; kc < 4; ++kc) {              // K = 128 ci in 4 chunks of 32
        int ci0 = kc * 32;
        v16h af;                                    // A: 16 co x 32 ci (contiguous ci)
        int m = lane & 15;
        const float* wrow = wp + ((size_t)tap * kCd + co0 + m) * kCin + ci0;
        #pragma unroll
        for (int i = 0; i < 16; ++i) af[i] = (_Float16)wrow[fragK(i, lane)];
        v16h bf;                                    // B: 32 ci x 16 pixels
        const float* xcol = x + (((size_t)b * kCin + ci0) * kH + hh) * kW + wcc;
        #pragma unroll
        for (int i = 0; i < 16; ++i) {
          float v = xcol[(size_t)fragK(i, lane) * kN];
          bf[i] = (_Float16)(v * msk);
        }
        acc = __builtin_amdgcn_wmma_f32_16x16x32_f16(false, af, false, bf, (short)0, acc, false, false);
      }
    }
  }
  int n = lane & 15;
  #pragma unroll
  for (int r = 0; r < 8; ++r) {                     // C/D layout: M=r(+8), N=lane&15
    int m  = r + ((lane >> 4) << 3);
    int co = co0 + m;
    float sc = g[co] * rsqrtf(va[co] + kBnEps);
    float sh = be[co] - mu[co] * sc;
    float v  = gelu_exact(acc[r] * sc + sh);
    feat[(((size_t)b * kCd + co) * kH + h) * kW + (w0 + n)] = v;
  }
}

// ---- DFT/iDFT stages as 128x128x128 GEMMs (split-bf16 WMMA) --------------
// MODE 0: rows fwd (real in):   OutR = A@Wr,        OutI = A@Wi
// MODE 1: cols fwd (cplx in):   OutR = Wr@Ar-Wi@Ai, OutI = Wr@Ai+Wi@Ar
// MODE 2: cols inv:             OutR = Wr@Ar+Wi@Ai, OutI = Wr@Ai-Wi@Ar
// MODE 3: rows inv + |.|/N + x: Ur = Ar@Wr+Ai@Wi,   Ui = Ai@Wr-Ar@Wi
// W is symmetric (W == W^T, conj(W)^T == conj(W)), so every W fragment is
// addressed row-major (contiguous k) regardless of which side it sits on.
// Modes 1/2 stage the complex image in padded LDS for conflict-free
// column-major fragment reads.
template <int MODE>
__global__ void dft_gemm(const float* __restrict__ inR, const float* __restrict__ inI,
                         const float* __restrict__ Wr, const float* __restrict__ Wi,
                         float* __restrict__ outR, float* __restrict__ outI,
                         const float* __restrict__ xin, float* __restrict__ yout, int chOff) {
  extern __shared__ float smem[];
  int img  = blockIdx.x;
  int wave = threadIdx.x >> 5;   // 8 waves, each owns one 16-row m-tile
  int lane = threadIdx.x & 31;
  int mt   = wave;
  const float* aR = inR + (size_t)img * kN;
  const float* aI = (MODE == 0) ? nullptr : (inI + (size_t)img * kN);
  __builtin_prefetch(aR + mt * 16 * kW, 0, 1);      // global_prefetch_b8

  if constexpr (MODE == 1 || MODE == 2) {
    // cooperative coalesced stage: global float4 -> padded LDS rows
    float* sR = smem;
    float* sI = smem + kH * kLdsPitch;
    const float4* gR = reinterpret_cast<const float4*>(aR);
    const float4* gI = reinterpret_cast<const float4*>(aI);
    for (int j = threadIdx.x; j < kN / 4; j += 256) {
      int r = j >> 5;                 // row (128 floats = 32 float4 per row)
      int c = (j & 31) << 2;          // col
      float4 vR = gR[j];
      float4 vI = gI[j];
      float* dR = &sR[r * kLdsPitch + c];
      float* dI = &sI[r * kLdsPitch + c];
      dR[0] = vR.x; dR[1] = vR.y; dR[2] = vR.z; dR[3] = vR.w;
      dI[0] = vI.x; dI[1] = vI.y; dI[2] = vI.z; dI[3] = vI.w;
    }
    __syncthreads();
  }

  for (int nt = 0; nt < 8; ++nt) {
    v8f accR = {};
    v8f accI = {};
    for (int kc = 0; kc < 4; ++kc) {
      int k0 = kc * 32;
      if constexpr (MODE == 0) {
        Frag a  = make_frag(aR + mt * 16 * kW + k0, kW, 1, lane);
        Frag wr = make_frag(Wr + nt * 16 * kW + k0, kW, 1, lane);  // symmetry
        Frag wi = make_frag(Wi + nt * 16 * kW + k0, kW, 1, lane);
        accR = mma3(a, wr, accR);
        accI = mma3(a, wi, accI);
      } else if constexpr (MODE == 1 || MODE == 2) {
        Frag fwr = make_frag(Wr + mt * 16 * kW + k0, kW, 1, lane);
        Frag fwi = make_frag(Wi + mt * 16 * kW + k0, kW, 1, lane);
        const float* sR = smem;
        const float* sI = smem + kH * kLdsPitch;
        Frag br = make_frag(sR + k0 * kLdsPitch + nt * 16, 1, kLdsPitch, lane);
        Frag bi = make_frag(sI + k0 * kLdsPitch + nt * 16, 1, kLdsPitch, lane);
        accR = mma3(fwr, br, accR);
        accI = mma3(fwr, bi, accI);
        if constexpr (MODE == 1) {
          Frag nbi = fneg(bi);
          accR = mma3(fwi, nbi, accR);
          accI = mma3(fwi, br, accI);
        } else {
          Frag nbr = fneg(br);
          accR = mma3(fwi, bi, accR);
          accI = mma3(fwi, nbr, accI);
        }
      } else {  // MODE 3
        Frag ar = make_frag(aR + mt * 16 * kW + k0, kW, 1, lane);
        Frag ai = make_frag(aI + mt * 16 * kW + k0, kW, 1, lane);
        Frag wr = make_frag(Wr + nt * 16 * kW + k0, kW, 1, lane);  // symmetry
        Frag wi = make_frag(Wi + nt * 16 * kW + k0, kW, 1, lane);
        accR = mma3(ar, wr, accR);
        accR = mma3(ai, wi, accR);
        accI = mma3(ai, wr, accI);
        Frag nwi = fneg(wi);
        accI = mma3(ar, nwi, accI);
      }
    }
    int n = lane & 15;
    #pragma unroll
    for (int r = 0; r < 8; ++r) {
      int m = r + ((lane >> 4) << 3);
      size_t pix = (size_t)(mt * 16 + m) * kW + (nt * 16 + n);
      if constexpr (MODE == 3) {
        float ur = accR[r] * (1.0f / (float)kN);
        float ui = accI[r] * (1.0f / (float)kN);
        int b = img >> 6;
        int c = img & 63;
        size_t gi = ((size_t)b * 128 + chOff + c) * kN + pix;
        yout[gi] = sqrtf(ur * ur + ui * ui) + xin[gi];   // fused abs + residual
      } else {
        outR[(size_t)img * kN + pix] = accR[r];
        outI[(size_t)img * kN + pix] = accI[r];
      }
    }
  }
}

// ---- Attention stats: complex Gram + norms + complex softmax per (b,head)
__global__ void attn_stats(const float* __restrict__ Fr, const float* __restrict__ Fi,
                           const float* __restrict__ tau,
                           float* __restrict__ aR, float* __restrict__ aI) {
  int bh = blockIdx.x;
  int b = bh >> 3, head = bh & 7;
  int tid = threadIdx.x;
  int p = tid & 63, grp = tid >> 6;
  int d = p >> 3, e = p & 7;
  const float* rd = Fr + ((size_t)b * kCd + head * kHd + d) * kN;
  const float* id = Fi + ((size_t)b * kCd + head * kHd + d) * kN;
  const float* re = Fr + ((size_t)b * kCd + head * kHd + e) * kN;
  const float* ie = Fi + ((size_t)b * kCd + head * kHd + e) * kN;
  float sr = 0.f, si = 0.f;
  for (int n = grp; n < kN; n += 4) {               // z_d * conj(z_e)
    float xr = rd[n], xi = id[n], yr = re[n], yi = ie[n];
    sr += xr * yr + xi * yi;
    si += xi * yr - xr * yi;
  }
  __shared__ float SR[256], SI[256];
  SR[tid] = sr; SI[tid] = si;
  __syncthreads();
  __shared__ float Gr[64], Gi[64], nm[8];
  if (tid < 64) {
    Gr[tid] = SR[tid] + SR[tid + 64] + SR[tid + 128] + SR[tid + 192];
    Gi[tid] = SI[tid] + SI[tid + 64] + SI[tid + 128] + SI[tid + 192];
  }
  __syncthreads();
  if (tid < 8) nm[tid] = sqrtf(Gr[tid * 8 + tid] + 1e-6f);  // diag = sum|z|^2
  __syncthreads();
  if (tid < 8) {
    int dd = tid;
    float t = tau[head];
    float vr[8], vi[8];
    #pragma unroll
    for (int ee = 0; ee < 8; ++ee) {
      float inv = t / (nm[dd] * nm[ee]);
      vr[ee] = Gr[dd * 8 + ee] * inv;
      vi[ee] = Gi[dd * 8 + ee] * inv;
    }
    float mr = vr[0], mi = vi[0];
    #pragma unroll
    for (int ee = 1; ee < 8; ++ee) { mr = fmaxf(mr, vr[ee]); mi = fmaxf(mi, vi[ee]); }
    float s1 = 0.f, s2 = 0.f;
    #pragma unroll
    for (int ee = 0; ee < 8; ++ee) {
      vr[ee] = expf(vr[ee] - mr); vi[ee] = expf(vi[ee] - mi);
      s1 += vr[ee]; s2 += vi[ee];
    }
    #pragma unroll
    for (int ee = 0; ee < 8; ++ee) {
      aR[(size_t)bh * 64 + dd * 8 + ee] = vr[ee] / s1;
      aI[(size_t)bh * 64 + dd * 8 + ee] = vi[ee] / s2;
    }
  }
}

// ---- Apply 8x8 complex attention over N (stream once, compute all d) -----
__global__ void attn_apply(const float* __restrict__ Fr, const float* __restrict__ Fi,
                           const float* __restrict__ aR, const float* __restrict__ aI,
                           float* __restrict__ Or, float* __restrict__ Oi) {
  int bh = blockIdx.x;
  int tid = threadIdx.x;
  __shared__ float sR[64], sI[64];
  if (tid < 64) { sR[tid] = aR[(size_t)bh * 64 + tid]; sI[tid] = aI[(size_t)bh * 64 + tid]; }
  __syncthreads();
  int n = blockIdx.y * 256 + tid;
  size_t base = ((size_t)(bh >> 3) * kCd + (bh & 7) * kHd) * kN;
  float vr[8], vi[8];
  #pragma unroll
  for (int e = 0; e < 8; ++e) {
    vr[e] = Fr[base + (size_t)e * kN + n];
    vi[e] = Fi[base + (size_t)e * kN + n];
  }
  #pragma unroll
  for (int d = 0; d < 8; ++d) {
    float orr = 0.f, oii = 0.f;
    #pragma unroll
    for (int e = 0; e < 8; ++e) {
      float ar = sR[d * 8 + e], ai = sI[d * 8 + e];
      orr += ar * vr[e] - ai * vi[e];
      oii += ar * vi[e] + ai * vr[e];
    }
    Or[base + (size_t)d * kN + n] = orr;
    Oi[base + (size_t)d * kN + n] = oii;
  }
}

// ---- Gate branch: sigmoid(W2·gelu(BN(W1·Re(F2)+b1))+b2) * F2 -------------
__global__ void gate_spectrum(const float* __restrict__ Fr, const float* __restrict__ Fi,
                              const float* __restrict__ w1, const float* __restrict__ b1,
                              const float* __restrict__ gg, const float* __restrict__ gb,
                              const float* __restrict__ gm, const float* __restrict__ gv,
                              const float* __restrict__ w2, const float* __restrict__ b2,
                              float* __restrict__ Gr, float* __restrict__ Gi) {
  int b = blockIdx.x;
  int n = blockIdx.y * 256 + threadIdx.x;
  size_t base = (size_t)b * kCd * kN;
  float mid[4];
  #pragma unroll
  for (int m = 0; m < 4; ++m) mid[m] = b1[m];
  for (int c = 0; c < kCd; ++c) {
    float fr = Fr[base + (size_t)c * kN + n];
    #pragma unroll
    for (int m = 0; m < 4; ++m) mid[m] += w1[m * kCd + c] * fr;
  }
  #pragma unroll
  for (int m = 0; m < 4; ++m) {
    float sc = gg[m] * rsqrtf(gv[m] + kBnEps);
    float sh = gb[m] - gm[m] * sc;
    mid[m] = gelu_exact(mid[m] * sc + sh);
  }
  for (int c = 0; c < kCd; ++c) {
    float a = b2[c];
    #pragma unroll
    for (int m = 0; m < 4; ++m) a += w2[c * 4 + m] * mid[m];
    float s = 1.0f / (1.0f + expf(-a));
    size_t idx = base + (size_t)c * kN + n;
    Gr[idx] = s * Fr[idx];
    Gi[idx] = s * Fi[idx];
  }
}

}  // namespace

extern "C" void kernel_launch(void* const* d_in, const int* in_sizes, int n_in,
                              void* d_out, int out_size, void* d_ws, size_t ws_size,
                              hipStream_t stream) {
  const float* x     = (const float*)d_in[0];
  const float* convw = (const float*)d_in[1];
  const float* bn1g  = (const float*)d_in[2];
  const float* bn1b  = (const float*)d_in[3];
  const float* bn1m  = (const float*)d_in[4];
  const float* bn1v  = (const float*)d_in[5];
  const float* cg1w  = (const float*)d_in[6];
  const float* cg1b  = (const float*)d_in[7];
  const float* cgbg  = (const float*)d_in[8];
  const float* cgbb  = (const float*)d_in[9];
  const float* cgbm  = (const float*)d_in[10];
  const float* cgbv  = (const float*)d_in[11];
  const float* cg2w  = (const float*)d_in[12];
  const float* cg2b  = (const float*)d_in[13];
  const float* tau   = (const float*)d_in[14];
  float* out = (float*)d_out;

  const size_t SZ = (size_t)kB * kCd * kN;   // one [B,64,H,W] f32 plane
  float* ws   = (float*)d_ws;
  float* feat = ws;                          // conv output
  float* Br   = ws + 1 * SZ;                 // row-DFT / attn-out / gate spectrum
  float* Bi   = ws + 2 * SZ;
  float* Cr   = ws + 3 * SZ;                 // F2 spectrum
  float* Ci   = ws + 4 * SZ;
  float* Dr   = ws + 5 * SZ;                 // inverse col-stage temp
  float* Di   = ws + 6 * SZ;
  float* Wr   = ws + 7 * SZ;                 // DFT twiddles 128x128
  float* Wi   = Wr + kN;
  float* aR   = Wi + kN;                     // attention weights [B*H, 8, 8]
  float* aI   = aR + (size_t)kB * kHeads * 64;
  float* wp   = aI + (size_t)kB * kHeads * 64;  // repacked conv weights [9][64][128]

  init_dft<<<kN / 256, 256, 0, stream>>>(Wr, Wi);
  repack_w<<<(9 * kCd * kCin + 255) / 256, 256, 0, stream>>>(convw, wp);

  // reduce branch: dilated conv + BN + exact GELU (f16 WMMA implicit GEMM)
  conv_bn_gelu<<<kB * kH * (kW / 16), 128, 0, stream>>>(x, wp, bn1g, bn1b, bn1m, bn1v, feat);

  // forward FFT2 as two GEMM stages (split-bf16 WMMA)
  dft_gemm<0><<<kB * kCd, 256, 0, stream>>>(feat, nullptr, Wr, Wi, Br, Bi, nullptr, nullptr, 0);
  dft_gemm<1><<<kB * kCd, 256, kLdsBytes, stream>>>(Br, Bi, Wr, Wi, Cr, Ci, nullptr, nullptr, 0);

  // frequency-domain attention
  attn_stats<<<kB * kHeads, 256, 0, stream>>>(Cr, Ci, tau, aR, aI);
  attn_apply<<<dim3(kB * kHeads, kN / 256), 256, 0, stream>>>(Cr, Ci, aR, aI, Br, Bi);

  // iFFT2 of attention output -> |.| + x -> out channels [0,64)
  dft_gemm<2><<<kB * kCd, 256, kLdsBytes, stream>>>(Br, Bi, Wr, Wi, Dr, Di, nullptr, nullptr, 0);
  dft_gemm<3><<<kB * kCd, 256, 0, stream>>>(Dr, Di, Wr, Wi, nullptr, nullptr, x, out, 0);

  // gated branch -> iFFT2 -> |.| + x -> out channels [64,128)
  gate_spectrum<<<dim3(kB, kN / 256), 256, 0, stream>>>(Cr, Ci, cg1w, cg1b, cgbg, cgbb, cgbm, cgbv,
                                                        cg2w, cg2b, Br, Bi);
  dft_gemm<2><<<kB * kCd, 256, kLdsBytes, stream>>>(Br, Bi, Wr, Wi, Dr, Di, nullptr, nullptr, 0);
  dft_gemm<3><<<kB * kCd, 256, 0, stream>>>(Dr, Di, Wr, Wi, nullptr, nullptr, x, out, kCd);

  (void)in_sizes; (void)n_in; (void)out_size; (void)ws_size;
}
